// HammingDistance_44306882626182
// MI455X (gfx1250) — 8-bit matrix operand order repaired and checked by reading, compiled, not run on hardware
//
#include <hip/hip_runtime.h>

typedef __attribute__((ext_vector_type(16))) int   v16i;
typedef __attribute__((ext_vector_type(8)))  float v8f;

#define N_ROWS  8192
#define D_TOK   512
#define K_TOK   32
#define K_STEPS 16
#define TILES   (N_ROWS / 16)

__device__ __forceinline__ int onehot(int t) { return (int)(0x38u << (t << 3)); }


__global__ __launch_bounds__(256) void encode_q_kernel(
    const int* __restrict__ xq, int4* __restrict__ aq)
{
  const int tid  = blockIdx.x * 256 + threadIdx.x;
  const int lane = tid & 31;
  const int kc   = (tid >> 5) & (K_STEPS - 1);
  const int rt   = tid >> 9;
  const int half = lane >> 4, l16 = lane & 15;
  const int* rp = xq + (size_t)(rt * 16 + l16) * D_TOK + kc * K_TOK;

  int w[16];
#pragma unroll
  for (int blk = 0; blk < 2; ++blk)
#pragma unroll
    for (int j = 0; j < 4; ++j) {
      const int2 t = *(const int2*)(rp + blk * 16 + 8 * (j >> 1) + 2 * (j & 1) + 4 * half);
      w[blk * 8 + 2 * j + 0] = onehot(t.x);
      w[blk * 8 + 2 * j + 1] = onehot(t.y);
    }
  int4* dst = aq + (size_t)tid * 4;
#pragma unroll
  for (int q = 0; q < 4; ++q)
    dst[q] = make_int4(w[4 * q], w[4 * q + 1], w[4 * q + 2], w[4 * q + 3]);
}

__global__ __launch_bounds__(256) void encode_p_kernel(
    const int* __restrict__ xp, int4* __restrict__ bp)
{
  const int tid  = blockIdx.x * 256 + threadIdx.x;
  const int lane = tid & 31;
  const int kc   = (tid >> 5) & (K_STEPS - 1);
  const int ct   = tid >> 9;
  const int half = lane >> 4, l16 = lane & 15;
  const int* rp = xp + (size_t)(ct * 16 + l16) * D_TOK + kc * K_TOK;

  int w[16];
#pragma unroll
  for (int g = 0; g < 4; ++g) {
    const int4 t = *(const int4*)(rp + g * 8 + 4 * half);
    w[g * 4 + 0] = onehot(t.x);
    w[g * 4 + 1] = onehot(t.y);
    w[g * 4 + 2] = onehot(t.z);
    w[g * 4 + 3] = onehot(t.w);
  }
  int4* dst = bp + (size_t)tid * 4;
#pragma unroll
  for (int q = 0; q < 4; ++q)
    dst[q] = make_int4(w[4 * q], w[4 * q + 1], w[4 * q + 2], w[4 * q + 3]);
}

__global__ __launch_bounds__(128) void hamming_wmma_fp8_pre(
    const int4* __restrict__ aq, const int4* __restrict__ bp,
    float* __restrict__ out)
{
  const int lane  = threadIdx.x & 31;
  const int wave  = threadIdx.x >> 5;
  const int waveR = wave >> 1, waveC = wave & 1;
  const int rowTile0 = blockIdx.y * 8 + waveR * 4;
  const int colTile0 = blockIdx.x * 8 + waveC * 4;
  const int half = lane >> 4, l16 = lane & 15;

  v8f acc[4][4];
#pragma unroll
  for (int r = 0; r < 4; ++r)
#pragma unroll
    for (int c = 0; c < 4; ++c)
      acc[r][c] = (v8f){0.f, 0.f, 0.f, 0.f, 0.f, 0.f, 0.f, 0.f};

  for (int kt = 0; kt < K_STEPS; ++kt) {
    v16i a[4], b[4];
#pragma unroll
    for (int r = 0; r < 4; ++r) {
      const int4* p =
          aq + (((size_t)(rowTile0 + r) * K_STEPS + kt) * 32 + lane) * 4;
      union { v16i v; int4 q[4]; } u;
      u.q[0] = p[0]; u.q[1] = p[1]; u.q[2] = p[2]; u.q[3] = p[3];
      a[r] = u.v;
    }
#pragma unroll
    for (int c = 0; c < 4; ++c) {
      const int4* p =
          bp + (((size_t)(colTile0 + c) * K_STEPS + kt) * 32 + lane) * 4;
      union { v16i v; int4 q[4]; } u;
      u.q[0] = p[0]; u.q[1] = p[1]; u.q[2] = p[2]; u.q[3] = p[3];
      b[c] = u.v;
    }
#pragma unroll
    for (int r = 0; r < 4; ++r)
#pragma unroll
      for (int c = 0; c < 4; ++c)
        acc[r][c] = __builtin_amdgcn_wmma_f32_16x16x128_fp8_fp8(
            a[r], b[c], (short)0, acc[r][c], false, false);
  }

#pragma unroll
  for (int r = 0; r < 4; ++r)
#pragma unroll
    for (int c = 0; c < 4; ++c) {
      const int n = (colTile0 + c) * 16 + l16;
#pragma unroll
      for (int i = 0; i < 8; ++i) {
        const int m = (rowTile0 + r) * 16 + 8 * half + i;
        __builtin_nontemporal_store(512.0f - acc[r][c][i],
                                    &out[(size_t)m * N_ROWS + n]);
      }
    }
}

__global__ __launch_bounds__(128) void hamming_wmma_fp8_fly(
    const int* __restrict__ xq, const int* __restrict__ xp,
    float* __restrict__ out)
{
  const int lane  = threadIdx.x & 31;
  const int wave  = threadIdx.x >> 5;
  const int waveR = wave >> 1, waveC = wave & 1;
  const int rowBase = blockIdx.y * 128 + waveR * 64;
  const int colBase = blockIdx.x * 128 + waveC * 64;
  const int half = lane >> 4, l16 = lane & 15;

  v8f acc[4][4];
#pragma unroll
  for (int r = 0; r < 4; ++r)
#pragma unroll
    for (int c = 0; c < 4; ++c)
      acc[r][c] = (v8f){0.f, 0.f, 0.f, 0.f, 0.f, 0.f, 0.f, 0.f};

  for (int kt = 0; kt < K_STEPS; ++kt) {
    const int kbase = kt * K_TOK;
    v16i a[4], b[4];
#pragma unroll
    for (int r = 0; r < 4; ++r) {
      const int* rp = xq + (size_t)(rowBase + r * 16 + l16) * D_TOK + kbase;
#pragma unroll
      for (int blk = 0; blk < 2; ++blk)
#pragma unroll
        for (int j = 0; j < 4; ++j) {
          const int2 t = *(const int2*)(rp + blk * 16 + 8 * (j >> 1) + 2 * (j & 1) + 4 * half);
          a[r][blk * 8 + 2 * j + 0] = onehot(t.x);
          a[r][blk * 8 + 2 * j + 1] = onehot(t.y);
        }
    }
#pragma unroll
    for (int c = 0; c < 4; ++c) {
      const int* pp = xp + (size_t)(colBase + c * 16 + l16) * D_TOK + kbase;
#pragma unroll
      for (int g = 0; g < 4; ++g) {
        const int4 t = *(const int4*)(pp + g * 8 + 4 * half);
        b[c][g * 4 + 0] = onehot(t.x);
        b[c][g * 4 + 1] = onehot(t.y);
        b[c][g * 4 + 2] = onehot(t.z);
        b[c][g * 4 + 3] = onehot(t.w);
      }
    }
#pragma unroll
    for (int r = 0; r < 4; ++r)
#pragma unroll
      for (int c = 0; c < 4; ++c)
        acc[r][c] = __builtin_amdgcn_wmma_f32_16x16x128_fp8_fp8(
            a[r], b[c], (short)0, acc[r][c], false, false);
  }

#pragma unroll
  for (int r = 0; r < 4; ++r)
#pragma unroll
    for (int c = 0; c < 4; ++c) {
      const int n = colBase + c * 16 + l16;
#pragma unroll
      for (int i = 0; i < 8; ++i) {
        const int m = rowBase + r * 16 + 8 * half + i;
        __builtin_nontemporal_store(512.0f - acc[r][c][i],
                                    &out[(size_t)m * N_ROWS + n]);
      }
    }
}

extern "C" void kernel_launch(void* const* d_in, const int* in_sizes, int n_in,
                              void* d_out, int out_size, void* d_ws, size_t ws_size,
                              hipStream_t stream) {
  const int* xq = (const int*)d_in[0];
  const int* xp = (const int*)d_in[1];
  float* out = (float*)d_out;

  const size_t enc_bytes = (size_t)TILES * K_STEPS * 32 * 64;

  if (ws_size >= 2 * enc_bytes) {
    int4* aq = (int4*)d_ws;
    int4* bp = (int4*)((char*)d_ws + enc_bytes);
    const int enc_threads = TILES * K_STEPS * 32;
    hipLaunchKernelGGL(encode_q_kernel, dim3(enc_threads / 256), dim3(256),
                       0, stream, xq, aq);
    hipLaunchKernelGGL(encode_p_kernel, dim3(enc_threads / 256), dim3(256),
                       0, stream, xp, bp);
    hipLaunchKernelGGL(hamming_wmma_fp8_pre, dim3(N_ROWS / 128, N_ROWS / 128),
                       dim3(128), 0, stream, aq, bp, out);
  } else {
    hipLaunchKernelGGL(hamming_wmma_fp8_fly, dim3(N_ROWS / 128, N_ROWS / 128),
                       dim3(128), 0, stream, xq, xp, out);
  }
}
